// AttentionAggregator_23751169147018
// MI455X (gfx1250) — matrix operand order repaired and checked by reading, compiled, not run on hardware
//
#include <hip/hip_runtime.h>

typedef __bf16 bf16;
typedef __attribute__((ext_vector_type(16))) __bf16 v16bf;
typedef __attribute__((ext_vector_type(8)))  float  v8f;

#define BSZ   20000
#define NN    32
#define DD    128
#define HH    256
#define OO    128
#define BT    16                 // nodes per block
#define ROWS  (BT*NN)            // 512 neighbor rows per block
#define LPAD  136                // padded bf16 row pitch (breaks LDS bank conflicts)
#define NKT   4                  // K tiles: 128 / 32

// ---------------------------------------------------------------------------
// Pre-swizzle a [128 x (nNt*16)] fp32 weight into exact WMMA B-fragment order:
// buf[((nt*NKT + kt)*32 + lane)*16 + i] = bf16( W[kt*32 + (i>=8?16:0) + (lane>=16?8:0) + (i&7)][nt*16 + (lane&15)] )
// so the main kernel loads one contiguous 32-byte v16bf per lane (fully coalesced).
// ---------------------------------------------------------------------------
__global__ __launch_bounds__(256) void swizzleW(const float* __restrict__ W,
                                                bf16* __restrict__ out, int nNt) {
  int idx = blockIdx.x * 256 + threadIdx.x;
  int total = nNt * NKT * 32 * 16;
  if (idx >= total) return;
  int i  = idx & 15;
  int L  = (idx >> 4) & 31;
  int kt = (idx >> 9) & 3;
  int nt = idx >> 11;
  int K  = kt * 32 + ((i >> 3) << 4) + ((L >> 4) << 3) + (i & 7);
  int n  = nt * 16 + (L & 15);
  out[idx] = (bf16)W[K * (nNt * 16) + n];
}

// ---------------------------------------------------------------------------
// Fused GAT block: scores (WMMA bf16) -> softmax -> aggregate -> output matmuls
// (WMMA bf16) -> relu -> store.  One block = 16 nodes, 8 waves.
// ---------------------------------------------------------------------------
__global__ __launch_bounds__(256) void gat_fused(
    const float* __restrict__ x,
    const float* __restrict__ neibs,
    const float* __restrict__ avec,      // a[0:256]=a1 (self), a[256:512]=a2 (neib)
    const bf16* __restrict__ Watt,       // swizzled [16][4][32][16]
    const bf16* __restrict__ Wfcx,       // swizzled [8][4][32][16]
    const bf16* __restrict__ Wfcn,       // swizzled [8][4][32][16]
    float* __restrict__ out)             // [B][256]
{
  __shared__ bf16  sh_neib[ROWS][LPAD];  // 139264 B
  __shared__ bf16  sh_x[BT][LPAD];
  __shared__ bf16  sh_agg[BT][LPAD];
  __shared__ float sh_s[BT][NN];
  __shared__ float sh_sx[BT];
  __shared__ float sh_att[BT][NN];
  __shared__ float sh_a[2 * HH];

  const int tid   = threadIdx.x;
  const int wave  = tid >> 5;
  const int lane  = tid & 31;
  const int lhalf = lane >> 4;
  const int lmod  = lane & 15;
  const int btile = blockIdx.x;

  // ---- Stage neighbor tile + x tile into LDS as bf16 (single global pass) ----
  {
    const float4* src = (const float4*)(neibs + (size_t)btile * ROWS * DD);
    for (int v = tid; v < ROWS * DD / 4; v += 256) {
      float4 f = src[v];
      int row = v >> 5;            // 32 float4 per row
      int col = (v & 31) << 2;
      sh_neib[row][col + 0] = (bf16)f.x;
      sh_neib[row][col + 1] = (bf16)f.y;
      sh_neib[row][col + 2] = (bf16)f.z;
      sh_neib[row][col + 3] = (bf16)f.w;
    }
    const float4* sx = (const float4*)(x + (size_t)btile * BT * DD);
    for (int v = tid; v < BT * DD / 4; v += 256) {
      float4 f = sx[v];
      int row = v >> 5;
      int col = (v & 31) << 2;
      sh_x[row][col + 0] = (bf16)f.x;
      sh_x[row][col + 1] = (bf16)f.y;
      sh_x[row][col + 2] = (bf16)f.z;
      sh_x[row][col + 3] = (bf16)f.w;
    }
    for (int v = tid; v < 2 * HH; v += 256) sh_a[v] = avec[v];
  }
  __syncthreads();

  // ---- Phase A: attention-score matmul, fused lrelu + dot(a) reduction ----
  // 33 row-groups of 16 rows (32 neighbor groups + 1 x group) across 8 waves.
  for (int g = wave; g < 33; g += 8) {
    const bf16* arow = (g < 32) ? &sh_neib[g * 16 + lmod][0] : &sh_x[lmod][0];

    // A fragments per ISA 16-bit A layout (lanes 0-15: M, K low/high halves).
    v16bf af[NKT];
#pragma unroll
    for (int kt = 0; kt < NKT; ++kt) {
      v16bf a;
#pragma unroll
      for (int i = 0; i < 16; ++i) {
        int v = i >> 1;
        int K = kt * 32 + ((v & 4) << 2) + ((v & 3) << 1) + (lhalf << 3) + (i & 1);
        a[i] = arow[K];
      }
      af[kt] = a;
    }

    float acc[8];
#pragma unroll
    for (int d = 0; d < 8; ++d) acc[d] = 0.f;

    const float* aw = (g < 32) ? (sh_a + HH) : sh_a;
    for (int nt = 0; nt < HH / 16; ++nt) {
      v8f c = {};
#pragma unroll
      for (int kt = 0; kt < NKT; ++kt) {
        v16bf b = *(const v16bf*)(Watt + (((nt * NKT + kt) * 32 + lane) << 4));
        c = __builtin_amdgcn_wmma_f32_16x16x32_bf16(false, af[kt], false, b,
                                                    (short)0, c, false, false);
      }
      float w = aw[nt * 16 + lmod];
#pragma unroll
      for (int d = 0; d < 8; ++d) {
        float vv = c[d];
        vv = vv > 0.f ? vv : 0.01f * vv;   // LeakyReLU(0.01) inside self.att
        acc[d] += vv * w;                  // fold dot with a-chunk immediately
      }
    }
    // Reduce over the 16 lanes holding the N dimension (within each half-wave).
#pragma unroll
    for (int d = 0; d < 8; ++d) {
      float vv = acc[d];
      vv += __shfl_xor(vv, 1, 32);
      vv += __shfl_xor(vv, 2, 32);
      vv += __shfl_xor(vv, 4, 32);
      vv += __shfl_xor(vv, 8, 32);
      acc[d] = vv;
    }
    if (lmod == 0) {
#pragma unroll
      for (int d = 0; d < 8; ++d) {
        int M = d + (lhalf << 3);
        if (g < 32) {
          int row = g * 16 + M;
          sh_s[row >> 5][row & 31] = acc[d];
        } else {
          sh_sx[M] = acc[d];
        }
      }
    }
  }
  __syncthreads();

  // ---- Phase B: softmax over neighbors: one wave32 per node, lane = neighbor ----
  for (int b = wave; b < BT; b += 8) {
    float e = sh_sx[b] + sh_s[b][lane];
    e = e > 0.f ? e : 0.2f * e;            // LeakyReLU(0.2) on scores
    float m = e;
    for (int off = 16; off; off >>= 1) m = fmaxf(m, __shfl_xor(m, off, 32));
    float ex = __expf(e - m);
    float sum = ex;
    for (int off = 16; off; off >>= 1) sum += __shfl_xor(sum, off, 32);
    sh_att[b][lane] = ex / sum;
  }
  __syncthreads();

  // ---- Phase C: attention-weighted aggregation from the LDS neighbor tile ----
  {
    int b  = tid >> 4;
    int dg = (tid & 15) << 3;              // 8 features per thread
    float accv[8];
#pragma unroll
    for (int d = 0; d < 8; ++d) accv[d] = 0.f;
    for (int n = 0; n < NN; ++n) {
      float at = sh_att[b][n];
      const bf16* rp = &sh_neib[b * NN + n][dg];
#pragma unroll
      for (int d = 0; d < 8; ++d) accv[d] += at * (float)rp[d];
    }
#pragma unroll
    for (int d = 0; d < 8; ++d) sh_agg[b][dg + d] = (bf16)accv[d];
  }
  __syncthreads();

  // ---- Phase D: output matmuls (x@Wfcx | agg@Wfcn), relu, store [B,256] ----
  {
    const bool  isX  = wave < 4;
    const bf16* arow = isX ? &sh_x[lmod][0] : &sh_agg[lmod][0];
    const bf16* Wb   = isX ? Wfcx : Wfcn;
    const int   coff = isX ? 0 : OO;
    const int   w2   = isX ? wave : wave - 4;

    v16bf af[NKT];
#pragma unroll
    for (int kt = 0; kt < NKT; ++kt) {
      v16bf a;
#pragma unroll
      for (int i = 0; i < 16; ++i) {
        int v = i >> 1;
        int K = kt * 32 + ((v & 4) << 2) + ((v & 3) << 1) + (lhalf << 3) + (i & 1);
        a[i] = arow[K];
      }
      af[kt] = a;
    }
#pragma unroll
    for (int t = 0; t < 2; ++t) {
      int nt = w2 * 2 + t;
      v8f c = {};
#pragma unroll
      for (int kt = 0; kt < NKT; ++kt) {
        v16bf b = *(const v16bf*)(Wb + (((nt * NKT + kt) * 32 + lane) << 4));
        c = __builtin_amdgcn_wmma_f32_16x16x32_bf16(false, af[kt], false, b,
                                                    (short)0, c, false, false);
      }
#pragma unroll
      for (int d = 0; d < 8; ++d) {
        int M   = d + (lhalf << 3);
        int gb  = btile * BT + M;
        int col = coff + nt * 16 + lmod;
        out[(size_t)gb * (2 * OO) + col] = fmaxf(c[d], 0.f);
      }
    }
  }
}

// ---------------------------------------------------------------------------
extern "C" void kernel_launch(void* const* d_in, const int* in_sizes, int n_in,
                              void* d_out, int out_size, void* d_ws, size_t ws_size,
                              hipStream_t stream) {
  const float* x     = (const float*)d_in[0];
  const float* neibs = (const float*)d_in[1];
  const float* W_att = (const float*)d_in[2];
  const float* W_fcx = (const float*)d_in[3];
  const float* W_fcn = (const float*)d_in[4];
  const float* a     = (const float*)d_in[5];
  float* out = (float*)d_out;

  bf16* ws_att = (bf16*)d_ws;            // 32768 elem
  bf16* ws_fcx = ws_att + DD * HH;       // 16384 elem
  bf16* ws_fcn = ws_fcx + DD * OO;       // 16384 elem

  swizzleW<<<(DD * HH + 255) / 256, 256, 0, stream>>>(W_att, ws_att, HH / 16);
  swizzleW<<<(DD * OO + 255) / 256, 256, 0, stream>>>(W_fcx, ws_fcx, OO / 16);
  swizzleW<<<(DD * OO + 255) / 256, 256, 0, stream>>>(W_fcn, ws_fcn, OO / 16);

  gat_fused<<<BSZ / BT, 256, 0, stream>>>(x, neibs, a, ws_att, ws_fcx, ws_fcn, out);
}
